// SynBlock_14199161881221
// MI455X (gfx1250) — matrix operand order repaired and checked by reading, not compiled, not run on hardware
//
#include <hip/hip_runtime.h>
#include <hip/hip_bf16.h>

typedef _Float16 half_t;
typedef __attribute__((ext_vector_type(16))) _Float16 v16h;
typedef __attribute__((ext_vector_type(8)))  float    v8f;

// ---------------------------------------------------------------------------
// gfx1250 async global->LDS copy (16B per lane), ASYNCcnt-tracked.
// Low 32 bits of a flat pointer into the shared aperture are the LDS address.
// ---------------------------------------------------------------------------
__device__ __forceinline__ void async_copy16(void* lds_dst, const void* gsrc) {
  uint32_t lds = (uint32_t)(size_t)lds_dst;
  asm volatile("global_load_async_to_lds_b128 %0, %1, off"
               :: "v"(lds), "v"((uint64_t)(size_t)gsrc) : "memory");
}
__device__ __forceinline__ void async_wait0() {
  asm volatile("s_wait_asynccnt 0x0" ::: "memory");
}

// ---------------------------------------------------------------------------
// Packing kernels
// ---------------------------------------------------------------------------

// fea [4,64,4,128,128] f32 -> x16 [16][128][128][64] f16 (img=b*4+t)
//                          -> fusein [4][128][128][256] f16 (ch=c*4+t)
__global__ __launch_bounds__(256) void k_pack_fea(const float* __restrict__ fea,
                                                  half_t* __restrict__ x16,
                                                  half_t* __restrict__ fusein) {
  int idx = blockIdx.x * 256 + threadIdx.x;          // 16,777,216 total
  int x = idx & 127; int rest = idx >> 7;
  int y = rest & 127; rest >>= 7;
  int t = rest & 3;  rest >>= 2;
  int c = rest & 63; int b = rest >> 6;
  half_t h = (half_t)fea[idx];
  x16[((((size_t)(b * 4 + t) * 128 + y) * 128 + x) * 64) + c] = h;
  fusein[(((size_t)b * 128 + y) * 128 + x) * 256 + (c * 4 + t)] = h;
}

// OIHW (or OI for 1x1) f32 -> [CoP][NTAP][CiP] f16, zero padded
template <int CoP, int CiP, int NTAP>
__global__ __launch_bounds__(256) void k_pack_convw(const float* __restrict__ w,
                                                    half_t* __restrict__ dst,
                                                    int CoV, int CiV) {
  int idx = blockIdx.x * 256 + threadIdx.x;
  if (idx >= CoP * NTAP * CiP) return;
  int ci = idx & (CiP - 1); int rest = idx / CiP;
  int tap = rest % NTAP; int oc = rest / NTAP;
  float val = 0.f;
  if (oc < CoV && ci < CiV)
    val = (NTAP == 9) ? w[(size_t)(oc * CiV + ci) * 9 + tap] : w[(size_t)oc * CiV + ci];
  dst[idx] = (half_t)val;
}

// ConvTranspose weight [CinV][CoutV][3][3] f32 -> per output-parity packed
// [4][CP][4][CP] f16. tap enumeration: ky asc x kx asc among valid taps.
template <int CP>
__global__ __launch_bounds__(256) void k_pack_deconvw(const float* __restrict__ w,
                                                      half_t* __restrict__ dst, int CV) {
  int idx = blockIdx.x * 256 + threadIdx.x;
  if (idx >= 4 * CP * 4 * CP) return;
  int ci = idx & (CP - 1); int rest = idx / CP;
  int tap = rest & 3; rest >>= 2;
  int oc = rest & (CP - 1); int pcc = rest / CP;
  int pyp = pcc >> 1, pxp = pcc & 1;
  int nky = pyp ? 2 : 1, nkx = pxp ? 2 : 1;
  float val = 0.f;
  if (tap < nky * nkx && oc < CV && ci < CV) {
    int iky = tap / nkx, ikx = tap % nkx;
    int ky = pyp ? (iky ? 2 : 0) : 1;
    int kx = pxp ? (ikx ? 2 : 0) : 1;
    val = w[((size_t)(ci * CV + oc) * 3 + ky) * 3 + kx];
  }
  dst[idx] = (half_t)val;
}

// ---------------------------------------------------------------------------
// WMMA implicit-GEMM conv (3x3 zero-pad or 1x1), NHWC f16, f32 accumulate.
// WG = 256 thr = 8 waves; wave -> 16 outch x 16 px tile; K = (tap, ci).
// A (weights, 16x32 f16): lane<16 rows hold K {0..7,16..23}, lane>=16 {8..15,24..31}
// B (input,  32x16 f16): lane<16 cols hold K {0..7,16..23}, lane>=16 {8..15,24..31}
// Input tile staged in LDS via global_load_async_to_lds_b128 (ASYNCcnt).
// ---------------------------------------------------------------------------
template <int CinP, int CoutP, int NTAP>
__global__ __launch_bounds__(256) void k_conv_wmma(
    const half_t* __restrict__ in, const half_t* __restrict__ wpk,
    const float* __restrict__ bias, half_t* __restrict__ out,
    int sshift /* log2(H)==log2(W) */, int CoutV, int lrelu) {
  constexpr int NG = CoutP >> 4;
  constexpr int NPH = 8 / NG;
  constexpr int WT = 16 * NPH;
  constexpr int WTS = (WT == 32) ? 5 : (WT == 64) ? 6 : 7;
  constexpr int HALO = (NTAP == 9) ? 1 : 0;
  constexpr int ROWS = (NTAP == 9) ? 3 : 1;
  constexpr int TCOLS = WT + 2 * HALO;
  constexpr int NKB = CinP >> 5;
  constexpr int CIV8 = CinP >> 3;
  constexpr int NVEC = ROWS * TCOLS * CIV8;
  __shared__ half_t sIn[ROWS * TCOLS * CinP];

  const int tid = threadIdx.x;
  const int lane = tid & 31;
  const int wv = tid >> 5;
  const int HW = 1 << sshift;
  const int tprS = sshift - WTS;
  int blk = blockIdx.x;
  const int tx = blk & ((1 << tprS) - 1); blk >>= tprS;
  const int y = blk & (HW - 1);
  const int img = blk >> sshift;
  const int x0 = tx << WTS;

  // ---- async tile stage: interior via ASYNC-to-LDS, halo zeros via ds ----
  for (int v = tid; v < NVEC; v += 256) {
    int ci8 = v % CIV8; int rest = v / CIV8;
    int tc = rest % TCOLS; int ky = rest / TCOLS;
    int gx = x0 + tc - HALO;
    int gy = y + ky - HALO;
    half_t* ldst = &sIn[(ky * TCOLS + tc) * CinP + ci8 * 8];
    if ((unsigned)gx < (unsigned)HW && (unsigned)gy < (unsigned)HW) {
      async_copy16(ldst, in + ((size_t)((img << sshift) + gy) * HW + gx) * CinP + ci8 * 8);
    } else {
      *(float4*)ldst = make_float4(0.f, 0.f, 0.f, 0.f);
    }
  }
  async_wait0();
  __syncthreads();

  const int g = wv % NG;
  const int ph = wv / NG;
  const int ochB = g * 16;
  const int n = lane & 15;
  const int hi = (lane >> 4) & 1;
  v8f acc;
#pragma unroll
  for (int r = 0; r < 8; ++r) {
    int ch = ochB + r + hi * 8;
    acc[r] = (ch < CoutV) ? bias[ch] : 0.f;
  }
  const int aR0 = hi ? 8 : 0;                         // A run base
  const int bR0 = hi ? 8 : 0;                         // B run base
  const half_t* wrow = wpk + (size_t)(ochB + n) * NTAP * CinP;
#pragma unroll
  for (int tap = 0; tap < NTAP; ++tap) {
    constexpr int three = 3;
    int ky = (NTAP == 9) ? tap / three : 0;
    int kx = (NTAP == 9) ? tap % three : 0;
    int tc = 16 * ph + n + kx;                        // lane's input tile col
    const half_t* bbase = &sIn[(ky * TCOLS + tc) * CinP];
    const half_t* abase = wrow + tap * CinP;
#pragma unroll
    for (int cb = 0; cb < NKB; ++cb) {
      union { v16h v; float4 f[2]; } A, Bm;
      A.f[0] = *(const float4*)(abase + cb * 32 + aR0);
      A.f[1] = *(const float4*)(abase + cb * 32 + aR0 + 16);
      Bm.f[0] = *(const float4*)(bbase + cb * 32 + bR0);
      Bm.f[1] = *(const float4*)(bbase + cb * 32 + bR0 + 16);
      acc = __builtin_amdgcn_wmma_f32_16x16x32_f16(false, A.v, false, Bm.v,
                                                   (short)0, acc, false, false);
    }
  }
  half_t hv[8];
#pragma unroll
  for (int r = 0; r < 8; ++r) {
    float f = acc[r];
    if (lrelu) f = (f >= 0.f) ? f : 0.2f * f;
    hv[r] = (half_t)f;
  }
  int px = x0 + 16 * ph + n;
  half_t* op = out + ((size_t)((img << sshift) + y) * HW + px) * CoutP + ochB + hi * 8;
  *(float4*)op = *(const float4*)hv;
}

// ---------------------------------------------------------------------------
// WMMA stride-2 transposed conv (k3,p1,op1) via 4 output-parity GEMMs.
// in [nimg][h][w][CP] f16 -> out [nimg][2h][2w][CP] f16,  h==w==1<<sshift
// ---------------------------------------------------------------------------
template <int CP>
__global__ __launch_bounds__(256) void k_deconv_wmma(
    const half_t* __restrict__ in, const half_t* __restrict__ wpk,
    const float* __restrict__ bias, half_t* __restrict__ out,
    int sshift, int CV) {
  constexpr int NG = CP >> 4;
  constexpr int NPH = 8 / NG;
  constexpr int WT = 16 * NPH;
  constexpr int WTS = (WT == 32) ? 5 : 6;
  constexpr int CT = (WT >> 1) + 1;
  constexpr int NKB = CP >> 5;
  constexpr int CIV8 = CP >> 3;
  constexpr int NVEC = 2 * CT * CIV8;
  __shared__ half_t sIn[2 * CT * CP];

  const int tid = threadIdx.x, lane = tid & 31, wv = tid >> 5;
  const int h = 1 << sshift;
  const int oS = sshift + 1;                          // log2 of output dim
  const int tprS = oS - WTS;
  int blk = blockIdx.x;
  const int tx = blk & ((1 << tprS) - 1); blk >>= tprS;
  const int oy = blk & ((1 << oS) - 1);
  const int img = blk >> oS;
  const int x0 = tx << WTS;
  const int pyp = oy & 1;
  const int iy_lo = (oy - pyp) >> 1;
  const int ixb = x0 >> 1;

  for (int v = tid; v < NVEC; v += 256) {
    int ci8 = v % CIV8; int rest = v / CIV8;
    int c = rest % CT; int r = rest / CT;
    int iy = iy_lo + r, ix = ixb + c;
    half_t* ldst = &sIn[(r * CT + c) * CP + ci8 * 8];
    if ((unsigned)iy < (unsigned)h && (unsigned)ix < (unsigned)h) {
      async_copy16(ldst, in + ((size_t)((img << sshift) + iy) * h + ix) * CP + ci8 * 8);
    } else {
      *(float4*)ldst = make_float4(0.f, 0.f, 0.f, 0.f);
    }
  }
  async_wait0();
  __syncthreads();

  const int g = wv % NG, ph = wv / NG;
  const int pxp = ph & 1, sub = ph >> 1;
  const int ochB = g * 16, n = lane & 15, hi = (lane >> 4) & 1;
  const int ox = x0 + sub * 32 + 2 * n + pxp;
  const int pc = pyp * 2 + pxp;
  const int nky = pyp ? 2 : 1, nkx = pxp ? 2 : 1;
  v8f acc;
#pragma unroll
  for (int r = 0; r < 8; ++r) {
    int ch = ochB + r + hi * 8;
    acc[r] = (ch < CV) ? bias[ch] : 0.f;
  }
  const int aR0 = hi ? 8 : 0, bR0 = hi ? 8 : 0;
  const half_t* wrow = wpk + (size_t)((pc * CP + ochB + n) * 4) * CP;
  for (int iky = 0; iky < nky; ++iky) {
    int ky = pyp ? (iky ? 2 : 0) : 1;
    int tr = pyp ? ((ky == 0) ? 1 : 0) : 0;
    for (int ikx = 0; ikx < nkx; ++ikx) {
      int kx = pxp ? (ikx ? 2 : 0) : 1;
      int col = ((ox + 1 - kx) >> 1) - ixb;
      int tap = iky * nkx + ikx;
      const half_t* bbase = &sIn[(tr * CT + col) * CP];
      const half_t* abase = wrow + tap * CP;
#pragma unroll
      for (int cb = 0; cb < NKB; ++cb) {
        union { v16h v; float4 f[2]; } A, Bm;
        A.f[0] = *(const float4*)(abase + cb * 32 + aR0);
        A.f[1] = *(const float4*)(abase + cb * 32 + aR0 + 16);
        Bm.f[0] = *(const float4*)(bbase + cb * 32 + bR0);
        Bm.f[1] = *(const float4*)(bbase + cb * 32 + bR0 + 16);
        acc = __builtin_amdgcn_wmma_f32_16x16x32_f16(false, A.v, false, Bm.v,
                                                     (short)0, acc, false, false);
      }
    }
  }
  half_t hv[8];
#pragma unroll
  for (int r = 0; r < 8; ++r) hv[r] = (half_t)acc[r];
  half_t* op = out + ((size_t)((img << oS) + oy) * (2 * h) + ox) * CP + ochB + hi * 8;
  *(float4*)op = *(const float4*)hv;
}

// ---------------------------------------------------------------------------
// Channel softmax over NHWC f16 (channels contiguous per pixel), in place.
// ---------------------------------------------------------------------------
__global__ __launch_bounds__(256) void k_softmax(half_t* __restrict__ buf,
                                                 int total, int Cp, int Cv) {
  int idx = blockIdx.x * 256 + threadIdx.x;
  if (idx >= total) return;
  half_t* p = buf + (size_t)idx * Cp;
  float v[25];
  float m = -1e30f;
  for (int c = 0; c < Cv; ++c) { v[c] = (float)p[c]; m = fmaxf(m, v[c]); }
  float s = 0.f;
  for (int c = 0; c < Cv; ++c) { v[c] = __expf(v[c] - m); s += v[c]; }
  float inv = 1.f / s;
  for (int c = 0; c < Cv; ++c) p[c] = (half_t)(v[c] * inv);
}

// ---------------------------------------------------------------------------
// AdaCoF sampling + occlusion blend. frames [4][4][3][256][256] f32,
// occ [4][256][256][16] f16 (4 valid), maps [16][256][256][32] f16 (25 valid).
// out [4][3][256][256] f32.
// ---------------------------------------------------------------------------
__global__ __launch_bounds__(256) void k_adacof(const float* __restrict__ frames,
                                                const half_t* __restrict__ occ,
                                                const half_t* __restrict__ wm,
                                                const half_t* __restrict__ am,
                                                const half_t* __restrict__ bm,
                                                float* __restrict__ out) {
  int idx = blockIdx.x * 256 + threadIdx.x;           // 4*65536
  int ox = idx & 255; int rest = idx >> 8;
  int oy = rest & 255; int b = rest >> 8;
  float acc0 = 0.f, acc1 = 0.f, acc2 = 0.f;
  size_t pixbase = ((size_t)b * 256 + oy) * 256 + ox;
  for (int t = 0; t < 4; ++t) {
    float ow = (float)occ[pixbase * 16 + t];
    int img = b * 4 + t;
    size_t mbase = (((size_t)img * 256 + oy) * 256 + ox) * 32;
    union { float4 f4[4]; half_t h[32]; } Wv, Av, Bv;
#pragma unroll
    for (int q = 0; q < 4; ++q) {
      Wv.f4[q] = ((const float4*)(wm + mbase))[q];
      Av.f4[q] = ((const float4*)(am + mbase))[q];
      Bv.f4[q] = ((const float4*)(bm + mbase))[q];
    }
    const float* fr = frames + (size_t)(t * 4 + b) * 3 * 65536;
    float s0 = 0.f, s1 = 0.f, s2 = 0.f;
#pragma unroll 5
    for (int k = 0; k < 25; ++k) {
      float kw = (float)Wv.h[k];
      float al = (float)Av.h[k];
      float be = (float)Bv.h[k];
      int dy = k / 5, dx = k - dy * 5;
      float yc = (float)(oy + dy) + al;
      float xc = (float)(ox + dx) + be;
      float yf = floorf(yc), xf = floorf(xc);
      float fy = yc - yf, fx = xc - xf;
      int y0 = min(max((int)yf, 0), 259);
      int x0 = min(max((int)xf, 0), 259);
      int y1 = min(y0 + 1, 259), x1 = min(x0 + 1, 259);
      int r0 = min(max(y0 - 2, 0), 255), r1 = min(max(y1 - 2, 0), 255);
      int c0 = min(max(x0 - 2, 0), 255), c1 = min(max(x1 - 2, 0), 255);
      float w00 = (1.f - fy) * (1.f - fx), w01 = (1.f - fy) * fx;
      float w10 = fy * (1.f - fx), w11 = fy * fx;
      int i00 = r0 * 256 + c0, i01 = r0 * 256 + c1;
      int i10 = r1 * 256 + c0, i11 = r1 * 256 + c1;
      s0 += kw * (fr[i00] * w00 + fr[i01] * w01 + fr[i10] * w10 + fr[i11] * w11);
      const float* f1 = fr + 65536;
      s1 += kw * (f1[i00] * w00 + f1[i01] * w01 + f1[i10] * w10 + f1[i11] * w11);
      const float* f2 = fr + 131072;
      s2 += kw * (f2[i00] * w00 + f2[i01] * w01 + f2[i10] * w10 + f2[i11] * w11);
    }
    acc0 += ow * s0; acc1 += ow * s1; acc2 += ow * s2;
  }
  size_t ob = (size_t)b * 3 * 65536 + (size_t)oy * 256 + ox;
  out[ob] = acc0; out[ob + 65536] = acc1; out[ob + 131072] = acc2;
}

// ---------------------------------------------------------------------------
// Host orchestration
// ---------------------------------------------------------------------------
struct SubP { const float *w1, *b1, *w2, *b2, *wd, *bd, *w3, *b3; };

extern "C" void kernel_launch(void* const* d_in, const int* in_sizes, int n_in,
                              void* d_out, int out_size, void* d_ws, size_t ws_size,
                              hipStream_t stream) {
  auto F = [&](int i) { return (const float*)d_in[i]; };
  const float *fea, *frames, *fuse_w, *fuse_b;
  SubP sw, sa, sb, so;
  if (in_sizes[0] > 1) {
    // insertion order: fea, frames, fuse_w, fuse_b, weight{w1,b1,w2,b2,wd,bd,w3,b3}, alpha, beta, occ, H, W
    fea = F(0); frames = F(1); fuse_w = F(2); fuse_b = F(3);
    auto sub = [&](int i) { return SubP{F(i), F(i+1), F(i+2), F(i+3), F(i+4), F(i+5), F(i+6), F(i+7)}; };
    sw = sub(4); sa = sub(12); sb = sub(20); so = sub(28);
  } else {
    // jax pytree sorted: H, W, fea, frames, params{alpha,beta,fuse_b,fuse_w,occ,weight},
    // subnet leaves sorted: b1,b2,b3,bd,w1,w2,w3,wd
    fea = F(2); frames = F(3);
    auto sub = [&](int i) { return SubP{F(i+4), F(i), F(i+5), F(i+1), F(i+7), F(i+3), F(i+6), F(i+2)}; };
    sa = sub(4); sb = sub(12); fuse_b = F(20); fuse_w = F(21); so = sub(22); sw = sub(30);
  }

  char* wsb = (char*)d_ws;
  size_t off = 0;
  auto alloc = [&](size_t bytes) {
    off = (off + 255) & ~(size_t)255;
    void* p = wsb + off;
    off += bytes;
    return p;
  };
  half_t* x16    = (half_t*)alloc((size_t)16 * 128 * 128 * 64 * 2);
  half_t* fusein = (half_t*)alloc((size_t)4 * 128 * 128 * 256 * 2);
  half_t* occf   = (half_t*)alloc((size_t)4 * 128 * 128 * 64 * 2);
  half_t* t1     = (half_t*)alloc((size_t)16 * 128 * 128 * 64 * 2);
  half_t* t2     = (half_t*)alloc((size_t)16 * 128 * 128 * 32 * 2);
  half_t* t3     = (half_t*)alloc((size_t)16 * 256 * 256 * 32 * 2); // also occ deconv (4 img x 64ch)
  half_t* wmapp  = (half_t*)alloc((size_t)16 * 256 * 256 * 32 * 2);
  half_t* amap   = (half_t*)alloc((size_t)16 * 256 * 256 * 32 * 2);
  half_t* bmap   = (half_t*)alloc((size_t)16 * 256 * 256 * 32 * 2);
  half_t* occm   = (half_t*)alloc((size_t)4 * 256 * 256 * 16 * 2);
  half_t* fusepk = (half_t*)alloc((size_t)64 * 1 * 256 * 2);
  half_t* pk_w1[4]; half_t* pk_w2[4]; half_t* pk_wd[4]; half_t* pk_w3[4];
  for (int s = 0; s < 3; ++s) {                       // weight, alpha, beta
    pk_w1[s] = (half_t*)alloc((size_t)64 * 9 * 64 * 2);
    pk_w2[s] = (half_t*)alloc((size_t)32 * 9 * 64 * 2);
    pk_wd[s] = (half_t*)alloc((size_t)4 * 32 * 4 * 32 * 2);
    pk_w3[s] = (half_t*)alloc((size_t)32 * 9 * 32 * 2);
  }
  pk_w1[3] = (half_t*)alloc((size_t)64 * 9 * 64 * 2); // occ
  pk_w2[3] = (half_t*)alloc((size_t)64 * 9 * 64 * 2);
  pk_wd[3] = (half_t*)alloc((size_t)4 * 64 * 4 * 64 * 2);
  pk_w3[3] = (half_t*)alloc((size_t)16 * 9 * 64 * 2);

  auto cdiv = [](int a, int b) { return (a + b - 1) / b; };

  // --- packing ---
  k_pack_fea<<<65536, 256, 0, stream>>>(fea, x16, fusein);
  k_pack_convw<64, 256, 1><<<cdiv(64 * 256, 256), 256, 0, stream>>>(fuse_w, fusepk, 64, 256);
  const SubP* subs[4] = {&sw, &sa, &sb, &so};
  for (int s = 0; s < 3; ++s) {
    k_pack_convw<64, 64, 9><<<cdiv(64 * 9 * 64, 256), 256, 0, stream>>>(subs[s]->w1, pk_w1[s], 64, 64);
    k_pack_convw<32, 64, 9><<<cdiv(32 * 9 * 64, 256), 256, 0, stream>>>(subs[s]->w2, pk_w2[s], 25, 64);
    k_pack_deconvw<32><<<cdiv(4 * 32 * 4 * 32, 256), 256, 0, stream>>>(subs[s]->wd, pk_wd[s], 25);
    k_pack_convw<32, 32, 9><<<cdiv(32 * 9 * 32, 256), 256, 0, stream>>>(subs[s]->w3, pk_w3[s], 25, 25);
  }
  k_pack_convw<64, 64, 9><<<cdiv(64 * 9 * 64, 256), 256, 0, stream>>>(so.w1, pk_w1[3], 64, 64);
  k_pack_convw<64, 64, 9><<<cdiv(64 * 9 * 64, 256), 256, 0, stream>>>(so.w2, pk_w2[3], 64, 64);
  k_pack_deconvw<64><<<cdiv(4 * 64 * 4 * 64, 256), 256, 0, stream>>>(so.wd, pk_wd[3], 64);
  k_pack_convw<16, 64, 9><<<cdiv(16 * 9 * 64, 256), 256, 0, stream>>>(so.w3, pk_w3[3], 4, 64);

  // --- fuse 1x1 conv + lrelu: [4,256,128,128] -> occf [4][128][128][64] ---
  k_conv_wmma<256, 64, 1><<<4 * 128 * 4, 256, 0, stream>>>(fusein, fusepk, fuse_b, occf, 7, 64, 1);
  // --- occlusion subnet ---
  k_conv_wmma<64, 64, 9><<<4 * 128 * 4, 256, 0, stream>>>(occf, pk_w1[3], so.b1, t1, 7, 64, 1);
  k_conv_wmma<64, 64, 9><<<4 * 128 * 4, 256, 0, stream>>>(t1, pk_w2[3], so.b2, occf, 7, 64, 1);
  k_deconv_wmma<64><<<4 * 256 * 8, 256, 0, stream>>>(occf, pk_wd[3], so.bd, t3, 7, 64);
  k_conv_wmma<64, 16, 9><<<4 * 256 * 2, 256, 0, stream>>>(t3, pk_w3[3], so.b3, occm, 8, 4, 0);
  k_softmax<<<cdiv(4 * 65536, 256), 256, 0, stream>>>(occm, 4 * 65536, 16, 4);

  // --- weight / alpha / beta subnets (16 images each) ---
  half_t* maps[3] = {wmapp, amap, bmap};
  for (int s = 0; s < 3; ++s) {
    k_conv_wmma<64, 64, 9><<<16 * 128 * 4, 256, 0, stream>>>(x16, pk_w1[s], subs[s]->b1, t1, 7, 64, 1);
    k_conv_wmma<64, 32, 9><<<16 * 128 * 2, 256, 0, stream>>>(t1, pk_w2[s], subs[s]->b2, t2, 7, 25, 1);
    k_deconv_wmma<32><<<16 * 256 * 4, 256, 0, stream>>>(t2, pk_wd[s], subs[s]->bd, t3, 7, 25);
    k_conv_wmma<32, 32, 9><<<16 * 256 * 4, 256, 0, stream>>>(t3, pk_w3[s], subs[s]->b3, maps[s], 8, 25, 0);
  }
  k_softmax<<<cdiv(16 * 65536, 256), 256, 0, stream>>>(wmapp, 16 * 65536, 32, 25);

  // --- final AdaCoF blend ---
  k_adacof<<<cdiv(4 * 65536, 256), 256, 0, stream>>>(frames, occm, wmapp, amap, bmap,
                                                     (float*)d_out);
}
